// MLPRes_54348516163780
// MI455X (gfx1250) — matrix operand order repaired and checked by reading, not compiled, not run on hardware
//
#include <hip/hip_runtime.h>
#include <hip/hip_bf16.h>

#define HID   512
#define NBLK  20
#define NTOT  65536
#define MTILE 64
#define LN_EPS 1e-5f

typedef __attribute__((ext_vector_type(16))) __bf16 v16bf;
typedef __attribute__((ext_vector_type(8)))  float  v8f;

__device__ __forceinline__ unsigned short f2bf(float f) {
    unsigned int u = __float_as_uint(f);
    u += 0x7FFFu + ((u >> 16) & 1u);          // round-to-nearest-even
    return (unsigned short)(u >> 16);
}

__device__ __forceinline__ float silu_f(float v) {
    return v / (1.0f + __expf(-v));
}

// ---------------------------------------------------------------------------
// Pack fp32 weights into bf16 WMMA B-fragment order:
//   idx = ((((blk*2+g)*16 + kt)*32 + ntG)*32 + lane)*16 + e
//   B layout (16x16x32 bf16): lane L holds column n = ntG*16 + (L%16),
//   rows K = kt*32 + (L/16)*16 + e  (e = 0..15).
// ---------------------------------------------------------------------------
__global__ void wconv_kernel(const float* __restrict__ fc1,
                             const float* __restrict__ fc2,
                             unsigned short* __restrict__ wsbf) {
    int idx  = blockIdx.x * blockDim.x + threadIdx.x;
    int e    = idx & 15;
    int lane = (idx >> 4)  & 31;
    int ntG  = (idx >> 9)  & 31;
    int kt   = (idx >> 14) & 15;
    int g    = (idx >> 18) & 1;
    int b    = idx >> 19;
    int k = kt * 32 + (lane >> 4) * 16 + e;
    int n = ntG * 16 + (lane & 15);
    const float* W = g ? fc2 : fc1;
    wsbf[idx] = f2bf(W[((size_t)b * HID + k) * HID + n]);
}

// per-row (sum, sumsq) partials from a wave's 16x128 register chunk
__device__ __forceinline__ void row_stats(const v8f (&arr)[8], int lo,
                                          int rowLB, int nquad, float* red) {
    #pragma unroll
    for (int r = 0; r < 8; ++r) {
        float s = 0.f, q = 0.f;
        #pragma unroll
        for (int nt = 0; nt < 8; ++nt) { float v = arr[nt][r]; s += v; q += v * v; }
        #pragma unroll
        for (int m = 1; m < 16; m <<= 1) {
            s += __shfl_xor(s, m, 32);
            q += __shfl_xor(q, m, 32);
        }
        if (lo == 0) {
            int row = rowLB + r;
            red[(row * 4 + nquad) * 2 + 0] = s;
            red[(row * 4 + nquad) * 2 + 1] = q;
        }
    }
}

// one 16(M)x128(N)x512(K) GEMM slab per wave: A from LDS (row-major bf16),
// B fragments streamed from packed global weights (L2-resident).
__device__ __forceinline__ void gemm_tile(v8f (&acc)[8],
                                          const unsigned short* tBuf,
                                          const unsigned short* __restrict__ wfrag,
                                          int fragBase, int waveM, int nquad, int lane) {
    const int lo = lane & 15, hi = lane >> 4;
    #pragma unroll 2
    for (int kt = 0; kt < 16; ++kt) {
        // A fragment: one contiguous 32B run per lane (K = hi*16 + e), matching packed B
        union { uint4 u[2]; v16bf v; } au;
        const uint4* ap = (const uint4*)(tBuf + (waveM * 16 + lo) * HID + kt * 32 + hi * 16);
        au.u[0] = ap[0];   // K = hi*16 .. hi*16+7
        au.u[1] = ap[1];   // K = hi*16+8 .. hi*16+15
        #pragma unroll
        for (int nt = 0; nt < 8; ++nt) {
            int ntG = nquad * 8 + nt;
            const v16bf* bp = (const v16bf*)(wfrag +
                (((size_t)(fragBase + kt) * 32 + ntG) * 32 + lane) * 16);
            acc[nt] = __builtin_amdgcn_wmma_f32_16x16x32_bf16(
                false, au.v, false, *bp, (short)0, acc[nt], false, false);
        }
    }
}

__global__ __launch_bounds__(512)
void mlp_resnet_kernel(const float* __restrict__ x,
                       const float* __restrict__ in_w,
                       const float* __restrict__ in_b,
                       const float* __restrict__ ln1_g,
                       const float* __restrict__ ln1_b,
                       const float* __restrict__ fc1_b,
                       const float* __restrict__ ln2_g,
                       const float* __restrict__ ln2_b,
                       const float* __restrict__ fc2_b,
                       const float* __restrict__ oln_g,
                       const float* __restrict__ oln_b,
                       const float* __restrict__ out_w,
                       const float* __restrict__ out_b,
                       const unsigned short* __restrict__ wfrag,
                       float* __restrict__ out) {
    __shared__ unsigned short tBuf[MTILE * HID]; // 64 KB normalized bf16 activations
    __shared__ float red[MTILE * 4 * 2];         // per-quad (sum,sumsq) partials
    __shared__ float stat[MTILE * 2];            // per-row (mean, rstd)

    const int tid   = threadIdx.x;
    const int lane  = tid & 31;
    const int wave  = tid >> 5;
    const int waveM = wave >> 2;              // 0..3 : row group of 16
    const int nquad = wave & 3;               // 0..3 : 128-col slab
    const int lo = lane & 15, hi = lane >> 4;
    const int rowLB = waveM * 16 + hi * 8;    // + r (0..7) = local row (C/D layout)
    const int colB  = nquad * 128 + lo;       // + nt*16   = column
    const long gRow = (long)blockIdx.x * MTILE + rowLB;

    // ---- input projection + SiLU, straight into register-resident h ----
    v8f h[8];
    {
        float w0[8], w1[8], bb[8];
        #pragma unroll
        for (int nt = 0; nt < 8; ++nt) {
            int c = colB + nt * 16;
            w0[nt] = in_w[c]; w1[nt] = in_w[HID + c]; bb[nt] = in_b[c];
        }
        #pragma unroll
        for (int r = 0; r < 8; ++r) {
            float x0 = x[(gRow + r) * 2 + 0];
            float x1 = x[(gRow + r) * 2 + 1];
            #pragma unroll
            for (int nt = 0; nt < 8; ++nt)
                h[nt][r] = silu_f(x0 * w0[nt] + x1 * w1[nt] + bb[nt]);
        }
    }

    for (int blk = 0; blk < NBLK; ++blk) {
        // ================= LN1 =================
        row_stats(h, lo, rowLB, nquad, red);
        __syncthreads();
        if (tid < MTILE) {
            float S = 0.f, Q = 0.f;
            #pragma unroll
            for (int q4 = 0; q4 < 4; ++q4) { S += red[(tid*4+q4)*2]; Q += red[(tid*4+q4)*2+1]; }
            float mean = S * (1.f / HID);
            float var  = Q * (1.f / HID) - mean * mean;
            stat[tid*2] = mean; stat[tid*2+1] = rsqrtf(var + LN_EPS);
        }
        __syncthreads();
        {
            float gv[8], bv[8];
            #pragma unroll
            for (int nt = 0; nt < 8; ++nt) {
                int c = blk * HID + colB + nt * 16;
                gv[nt] = ln1_g[c]; bv[nt] = ln1_b[c];
            }
            #pragma unroll
            for (int r = 0; r < 8; ++r) {
                float mean = stat[(rowLB+r)*2], rs = stat[(rowLB+r)*2+1];
                #pragma unroll
                for (int nt = 0; nt < 8; ++nt)
                    tBuf[(rowLB+r)*HID + colB + nt*16] =
                        f2bf((h[nt][r] - mean) * rs * gv[nt] + bv[nt]);
            }
        }
        __syncthreads();

        // ================= GEMM1 + bias + SiLU =================
        v8f acc[8] = {};
        gemm_tile(acc, tBuf, wfrag, (blk * 2 + 0) * 16, waveM, nquad, lane);
        {
            float cv[8];
            #pragma unroll
            for (int nt = 0; nt < 8; ++nt) cv[nt] = fc1_b[blk * HID + colB + nt * 16];
            #pragma unroll
            for (int nt = 0; nt < 8; ++nt)
                #pragma unroll
                for (int r = 0; r < 8; ++r)
                    acc[nt][r] = silu_f(acc[nt][r] + cv[nt]);
        }

        // ================= LN2 =================
        row_stats(acc, lo, rowLB, nquad, red);
        __syncthreads();   // also orders GEMM1 tBuf reads vs t2 writes below
        if (tid < MTILE) {
            float S = 0.f, Q = 0.f;
            #pragma unroll
            for (int q4 = 0; q4 < 4; ++q4) { S += red[(tid*4+q4)*2]; Q += red[(tid*4+q4)*2+1]; }
            float mean = S * (1.f / HID);
            float var  = Q * (1.f / HID) - mean * mean;
            stat[tid*2] = mean; stat[tid*2+1] = rsqrtf(var + LN_EPS);
        }
        __syncthreads();
        {
            float gv[8], bv[8];
            #pragma unroll
            for (int nt = 0; nt < 8; ++nt) {
                int c = blk * HID + colB + nt * 16;
                gv[nt] = ln2_g[c]; bv[nt] = ln2_b[c];
            }
            #pragma unroll
            for (int r = 0; r < 8; ++r) {
                float mean = stat[(rowLB+r)*2], rs = stat[(rowLB+r)*2+1];
                #pragma unroll
                for (int nt = 0; nt < 8; ++nt)
                    tBuf[(rowLB+r)*HID + colB + nt*16] =
                        f2bf((acc[nt][r] - mean) * rs * gv[nt] + bv[nt]);
            }
        }
        __syncthreads();

        // ================= GEMM2 + bias + residual =================
        #pragma unroll
        for (int nt = 0; nt < 8; ++nt)
            #pragma unroll
            for (int r = 0; r < 8; ++r) acc[nt][r] = 0.f;
        gemm_tile(acc, tBuf, wfrag, (blk * 2 + 1) * 16, waveM, nquad, lane);
        {
            float cv[8];
            #pragma unroll
            for (int nt = 0; nt < 8; ++nt) cv[nt] = fc2_b[blk * HID + colB + nt * 16];
            #pragma unroll
            for (int nt = 0; nt < 8; ++nt)
                #pragma unroll
                for (int r = 0; r < 8; ++r)
                    h[nt][r] += acc[nt][r] + cv[nt];
        }
    }

    // ================= output head: silu(LN(h)) @ out_w + out_b =================
    row_stats(h, lo, rowLB, nquad, red);
    __syncthreads();
    if (tid < MTILE) {
        float S = 0.f, Q = 0.f;
        #pragma unroll
        for (int q4 = 0; q4 < 4; ++q4) { S += red[(tid*4+q4)*2]; Q += red[(tid*4+q4)*2+1]; }
        float mean = S * (1.f / HID);
        float var  = Q * (1.f / HID) - mean * mean;
        stat[tid*2] = mean; stat[tid*2+1] = rsqrtf(var + LN_EPS);
    }
    __syncthreads();
    {
        float gv[8], bv[8], wv[8];
        #pragma unroll
        for (int nt = 0; nt < 8; ++nt) {
            int c = colB + nt * 16;
            gv[nt] = oln_g[c]; bv[nt] = oln_b[c]; wv[nt] = out_w[c];
        }
        #pragma unroll
        for (int r = 0; r < 8; ++r) {
            float mean = stat[(rowLB+r)*2], rs = stat[(rowLB+r)*2+1];
            float p = 0.f;
            #pragma unroll
            for (int nt = 0; nt < 8; ++nt)
                p += silu_f((h[nt][r] - mean) * rs * gv[nt] + bv[nt]) * wv[nt];
            #pragma unroll
            for (int m = 1; m < 16; m <<= 1) p += __shfl_xor(p, m, 32);
            if (lo == 0) red[((rowLB + r) * 4 + nquad) * 2] = p;
        }
    }
    __syncthreads();
    if (tid < MTILE) {
        float S = out_b[0];
        #pragma unroll
        for (int q4 = 0; q4 < 4; ++q4) S += red[(tid * 4 + q4) * 2];
        out[(long)blockIdx.x * MTILE + tid] = S;
    }
}

extern "C" void kernel_launch(void* const* d_in, const int* in_sizes, int n_in,
                              void* d_out, int out_size, void* d_ws, size_t ws_size,
                              hipStream_t stream) {
    (void)in_sizes; (void)n_in; (void)out_size; (void)ws_size;
    const float* x      = (const float*)d_in[0];
    const float* in_w   = (const float*)d_in[1];
    const float* in_b   = (const float*)d_in[2];
    const float* ln1_g  = (const float*)d_in[3];
    const float* ln1_b  = (const float*)d_in[4];
    const float* fc1_w  = (const float*)d_in[5];
    const float* fc1_b  = (const float*)d_in[6];
    const float* ln2_g  = (const float*)d_in[7];
    const float* ln2_b  = (const float*)d_in[8];
    const float* fc2_w  = (const float*)d_in[9];
    const float* fc2_b  = (const float*)d_in[10];
    const float* oln_g  = (const float*)d_in[11];
    const float* oln_b  = (const float*)d_in[12];
    const float* out_w  = (const float*)d_in[13];
    const float* out_b  = (const float*)d_in[14];
    unsigned short* wsbf = (unsigned short*)d_ws;     // 20 MB bf16 packed weights

    // pack weights into WMMA B-fragment order (bf16), runs every call (deterministic)
    const int total = NBLK * 2 * 16 * 32 * 32 * 16;   // 10,485,760
    wconv_kernel<<<total / 256, 256, 0, stream>>>(fc1_w, fc2_w, wsbf);

    // persistent tile kernel: 1024 workgroups x 512 threads, 64 rows each
    mlp_resnet_kernel<<<NTOT / MTILE, 512, 0, stream>>>(
        x, in_w, in_b, ln1_g, ln1_b, fc1_b, ln2_g, ln2_b, fc2_b,
        oln_g, oln_b, out_w, out_b, wsbf, (float*)d_out);
}
